// ScaleAdaptiveDeformableAttn_28595892257675
// MI455X (gfx1250) — matrix operand order repaired and checked by reading, not compiled, not run on hardware
//
#include <hip/hip_runtime.h>

#define B_SZ 4
#define NQ   1024
#define DM   256
#define NH   8
#define HD   32
#define NIN  21760
#define MQ   (B_SZ * NQ)    // 4096
#define MV   (B_SZ * NIN)   // 87040

typedef __bf16 bf16_t;
typedef __attribute__((ext_vector_type(16))) __bf16 v16bf;
typedef __attribute__((ext_vector_type(8)))  __bf16 v8bf;
typedef __attribute__((ext_vector_type(8)))  float  v8f;

// ---------------------------------------------------------------- converts
__global__ void __launch_bounds__(256) cvt_f32_to_bf16(const float* __restrict__ in,
                                                       bf16_t* __restrict__ out, int n) {
  int i = blockIdx.x * blockDim.x + threadIdx.x;
  int stride = gridDim.x * blockDim.x;
  for (; i < n; i += stride) out[i] = (bf16_t)in[i];
}

// w[K,N] f32 (row-major) -> wT[N,K] bf16 so B-fragments are contiguous in K
__global__ void __launch_bounds__(256) transpose_cvt(const float* __restrict__ w,
                                                     bf16_t* __restrict__ wT,
                                                     int K, int N) {
  int t = blockIdx.x * blockDim.x + threadIdx.x;
  if (t >= K * N) return;
  int n = t / K, k = t - n * K;
  wT[t] = (bf16_t)w[(size_t)k * N + n];
}

// ---------------------------------------------------------------- WMMA GEMM
// C[M,N] = A[M,K](bf16) * BT[N,K](bf16)^T + bias, f32 out, optional ReLU.
// Block = 8 waves; wave w owns rows [ (bx*8+w)*16, +16 ), cols [ by*NT*16, +NT*16 ).
template <int NT, bool RELU>
__global__ void __launch_bounds__(256) gemm_bf16_wmma(const bf16_t* __restrict__ A,
                                                      const bf16_t* __restrict__ BT,
                                                      const float* __restrict__ bias,
                                                      float* __restrict__ C,
                                                      int M, int N, int K) {
  const int lane = threadIdx.x & 31;
  const int wave = threadIdx.x >> 5;
  const int m0   = (blockIdx.x * 8 + wave) * 16;
  const int n0   = blockIdx.y * (NT * 16);
  const int mr   = lane & 15;       // A: row within tile
  const int half = lane >> 4;       // A: K-half selector / B: K-half selector
  const int nc   = n0 + (lane & 15);// C/B: column for this lane (t adds 16*t)

  v8f acc[NT];
#pragma unroll
  for (int t = 0; t < NT; ++t) {
    float bv = bias[nc + t * 16];
    v8f a0 = {bv, bv, bv, bv, bv, bv, bv, bv};
    acc[t] = a0;
  }

  // A frag as loaded here (16x32, contiguous K, same order as B): lane(l<16)=row m, elems e: K=16*half+e
  const bf16_t* ap = A + (size_t)(m0 + mr) * K + half * 16;
  for (int k0 = 0; k0 < K; k0 += 32) {
    v8bf alo = *(const v8bf*)(ap + k0);        // K = k0 + 16h + 0..7
    v8bf ahi = *(const v8bf*)(ap + k0 + 8);    // K = k0 + 16h + 8..15
    v16bf a = __builtin_shufflevector(alo, ahi, 0, 1, 2, 3, 4, 5, 6, 7,
                                      8, 9, 10, 11, 12, 13, 14, 15);
    __builtin_prefetch(ap + k0 + 64, 0, 1);    // global_prefetch_b8
#pragma unroll
    for (int t = 0; t < NT; ++t) {
      // B frag (32x16): lane -> col n, K = 16*half + e  -> contiguous 32B in wT[n,:]
      const bf16_t* bp = BT + (size_t)(nc + t * 16) * K + k0 + half * 16;
      v16bf b = *(const v16bf*)bp;
      acc[t] = __builtin_amdgcn_wmma_f32_16x16x32_bf16(false, a, false, b, (short)0,
                                                       acc[t], false, false);
    }
  }

  // C/D layout: VGPR r, lane -> row m0 + r + 8*(lane>=16), col nc + 16*t
#pragma unroll
  for (int t = 0; t < NT; ++t) {
#pragma unroll
    for (int r = 0; r < 8; ++r) {
      float v = acc[t][r];
      if (RELU) v = fmaxf(v, 0.0f);
      C[(size_t)(m0 + r + 8 * half) * N + nc + t * 16] = v;
    }
  }
}

// ---------------------------------------------------------------- scale head
// s2 = sigmoid( relu_s1 @ w_scale2 + b_scale2 ), also written to d_out tail.
__global__ void __launch_bounds__(256) scale2_kernel(const float* __restrict__ s1,
                                                     const float* __restrict__ w2,
                                                     const float* __restrict__ b2,
                                                     float* __restrict__ s2,
                                                     float* __restrict__ out_scale,
                                                     int M) {
  int i = blockIdx.x * blockDim.x + threadIdx.x;
  if (i >= M) return;
  const float* r = s1 + (size_t)i * 64;
  float acc = b2[0];
#pragma unroll
  for (int j = 0; j < 64; ++j) acc += r[j] * w2[j];
  float sig = 1.0f / (1.0f + expf(-acc));
  s2[i] = sig;
  out_scale[i] = sig;
}

// ---------------------------------------------------------------- sampling
// One wave per (b, q, head). Lanes 0..15: one (level,point) sample each
// (offsets, locations, 16-wide softmax). Then all 32 lanes (= head dim) do the
// bilinear gather with per-sample params broadcast via shfl.
__global__ void __launch_bounds__(256) deform_sample(const float* __restrict__ value,   // [B,NIN,256] f32
                                                     const float* __restrict__ off,     // [MQ,256]
                                                     const float* __restrict__ attn,    // [MQ,128]
                                                     const float* __restrict__ s2,      // [MQ]
                                                     const float* __restrict__ refp,    // [B,NQ,4,2]
                                                     const int* __restrict__ shapes,    // [4][2] (h,w)
                                                     const int* __restrict__ lsi,       // [4]
                                                     bf16_t* __restrict__ hout) {       // [MQ,256] bf16
  const int lane = threadIdx.x & 31;
  const int wave = threadIdx.x >> 5;
  const int gid  = blockIdx.x * 8 + wave;   // (b*NQ+q)*8 + h
  const int h    = gid & 7;
  const int row  = gid >> 3;                // b*NQ + q
  const int b    = row >> 10;               // NQ == 1024

  // --- per-sample params on lanes 0..15 (lanes 16..31 mirror, unused) ---
  const int s   = lane & 15;                // s = lvl*4 + pt
  const int lvl = s >> 2;
  const int Hs  = shapes[lvl * 2 + 0];
  const int Ws  = shapes[lvl * 2 + 1];
  const float rx = refp[((size_t)row * 4 + lvl) * 2 + 0];
  const float ry = refp[((size_t)row * 4 + lvl) * 2 + 1];
  const float sc = s2[row] * 0.5f;          // MAX_OFFSET = 0.5
  const float* op = off + (size_t)row * 256 + h * 32 + s * 2;
  const float ox = tanhf(op[0]) * sc;
  const float oy = tanhf(op[1]) * sc;
  // loc = ref + off/normalizer(w,h);  g = loc*{W,H} - 0.5
  const float gx = (rx + ox / (float)Ws) * (float)Ws - 0.5f;
  const float gy = (ry + oy / (float)Hs) * (float)Hs - 0.5f;
  const float x0 = floorf(gx), y0 = floorf(gy);
  const float fx = gx - x0,   fy = gy - y0;

  // softmax over the 16 samples of this head
  float logit = attn[(size_t)row * 128 + h * 16 + s];
  float mx = logit;
  for (int o = 8; o >= 1; o >>= 1) mx = fmaxf(mx, __shfl_xor(mx, o, 32));
  float e = expf(logit - mx);
  float se = e;
  for (int o = 8; o >= 1; o >>= 1) se += __shfl_xor(se, o, 32);
  const float aw = e / se;

  // --- gather: all 32 lanes = head dims, broadcast sample params ---
  float acc = 0.0f;
  const float* vbase = value + (size_t)b * NIN * 256 + h * 32 + lane;
#pragma unroll 1
  for (int si = 0; si < 16; ++si) {
    const float baw = __shfl(aw, si, 32);
    const float bx0 = __shfl(x0, si, 32), by0 = __shfl(y0, si, 32);
    const float bfx = __shfl(fx, si, 32), bfy = __shfl(fy, si, 32);
    const int bl = si >> 2;
    const int H2 = shapes[bl * 2 + 0], W2 = shapes[bl * 2 + 1];
    const int base = lsi[bl];
    const int xi0 = (int)bx0, yi0 = (int)by0;
    const float wx0 = 1.0f - bfx, wy0 = 1.0f - bfy;
#pragma unroll
    for (int dy = 0; dy < 2; ++dy) {
      const int yi = yi0 + dy;
      if (yi < 0 || yi >= H2) continue;
      const float wy = dy ? bfy : wy0;
#pragma unroll
      for (int dx = 0; dx < 2; ++dx) {
        const int xi = xi0 + dx;
        if (xi < 0 || xi >= W2) continue;
        const float w = baw * wy * (dx ? bfx : wx0);
        acc += w * vbase[(size_t)(base + yi * W2 + xi) * 256];
      }
    }
  }
  hout[(size_t)row * 256 + h * 32 + lane] = (bf16_t)acc;
}

// ---------------------------------------------------------------- launch
extern "C" void kernel_launch(void* const* d_in, const int* in_sizes, int n_in,
                              void* d_out, int out_size, void* d_ws, size_t ws_size,
                              hipStream_t stream) {
  const float* query  = (const float*)d_in[0];
  const float* refp   = (const float*)d_in[1];
  const float* inflat = (const float*)d_in[2];
  const int*   shapes = (const int*)d_in[3];
  const int*   lsi    = (const int*)d_in[4];
  const float* w_s1   = (const float*)d_in[5];
  const float* b_s1   = (const float*)d_in[6];
  const float* w_s2   = (const float*)d_in[7];
  const float* b_s2   = (const float*)d_in[8];
  const float* w_off  = (const float*)d_in[9];
  const float* b_off  = (const float*)d_in[10];
  const float* w_attn = (const float*)d_in[11];
  const float* b_attn = (const float*)d_in[12];
  const float* w_val  = (const float*)d_in[13];
  const float* b_val  = (const float*)d_in[14];
  const float* w_out  = (const float*)d_in[15];
  const float* b_out  = (const float*)d_in[16];
  float* out = (float*)d_out;                 // [MQ*256] main + [MQ] scale

  char* ws = (char*)d_ws;
  size_t wo = 0;
  auto alloc = [&](size_t bytes) -> char* {
    char* p = ws + wo;
    wo += (bytes + 255) & ~(size_t)255;
    return p;
  };
  bf16_t* qbf    = (bf16_t*)alloc((size_t)MQ * DM * 2);
  bf16_t* inbf   = (bf16_t*)alloc((size_t)MV * DM * 2);
  bf16_t* wTval  = (bf16_t*)alloc(256 * 256 * 2);
  bf16_t* wToff  = (bf16_t*)alloc(256 * 256 * 2);
  bf16_t* wTattn = (bf16_t*)alloc(128 * 256 * 2);
  bf16_t* wTs1   = (bf16_t*)alloc(64 * 256 * 2);
  bf16_t* wTout  = (bf16_t*)alloc(256 * 256 * 2);
  float*  valf   = (float*)alloc((size_t)MV * DM * 4);
  float*  s1f    = (float*)alloc((size_t)MQ * 64 * 4);
  float*  s2f    = (float*)alloc((size_t)MQ * 4);
  float*  offf   = (float*)alloc((size_t)MQ * 256 * 4);
  float*  attnf  = (float*)alloc((size_t)MQ * 128 * 4);
  bf16_t* houtb  = (bf16_t*)alloc((size_t)MQ * 256 * 2);

  // bf16 conversions
  cvt_f32_to_bf16<<<1024, 256, 0, stream>>>(query, qbf, MQ * DM);
  cvt_f32_to_bf16<<<4096, 256, 0, stream>>>(inflat, inbf, MV * DM);
  transpose_cvt<<<(256 * 256 + 255) / 256, 256, 0, stream>>>(w_val, wTval, 256, 256);
  transpose_cvt<<<(256 * 256 + 255) / 256, 256, 0, stream>>>(w_off, wToff, 256, 256);
  transpose_cvt<<<(128 * 256 + 255) / 256, 256, 0, stream>>>(w_attn, wTattn, 256, 128);
  transpose_cvt<<<(64 * 256 + 255) / 256, 256, 0, stream>>>(w_s1, wTs1, 256, 64);
  transpose_cvt<<<(256 * 256 + 255) / 256, 256, 0, stream>>>(w_out, wTout, 256, 256);

  // value projection (dominant GEMM): [87040,256] x [256,256]
  gemm_bf16_wmma<8, false><<<dim3(MV / 128, 2), 256, 0, stream>>>(
      inbf, wTval, b_val, valf, MV, 256, 256);

  // query-side projections
  gemm_bf16_wmma<4, true><<<dim3(MQ / 128, 1), 256, 0, stream>>>(
      qbf, wTs1, b_s1, s1f, MQ, 64, 256);   // fused ReLU
  scale2_kernel<<<MQ / 256, 256, 0, stream>>>(s1f, w_s2, b_s2, s2f,
                                              out + (size_t)MQ * DM, MQ);
  gemm_bf16_wmma<8, false><<<dim3(MQ / 128, 2), 256, 0, stream>>>(
      qbf, wToff, b_off, offf, MQ, 256, 256);
  gemm_bf16_wmma<8, false><<<dim3(MQ / 128, 1), 256, 0, stream>>>(
      qbf, wTattn, b_attn, attnf, MQ, 128, 256);

  // deformable bilinear gather + softmax-weighted sum -> per-head outputs
  deform_sample<<<MQ, 256, 0, stream>>>(valf, offf, attnf, s2f, refp, shapes, lsi, houtb);

  // output projection straight into d_out
  gemm_bf16_wmma<8, false><<<dim3(MQ / 128, 2), 256, 0, stream>>>(
      houtb, wTout, b_out, out, MQ, 256, 256);
}
